// FusedAttn_15058155339950
// MI455X (gfx1250) — matrix operand order repaired and checked by reading, not compiled, not run on hardware
//
#include <hip/hip_runtime.h>

// ---------------------------------------------------------------------------
// CDNA5 (gfx1250) fused entity-aware attention.
// All matrix contractions via v_wmma_f32_16x16x32_bf16 (bf16 in, f32 accum).
// Staging uses TDM tensor_load_to_lds / async global->LDS where available.
// ---------------------------------------------------------------------------

typedef __bf16 bf16_t;
typedef __attribute__((ext_vector_type(16))) __bf16 v16bf;
typedef __attribute__((ext_vector_type(8)))  __bf16 v8bf;
typedef __attribute__((ext_vector_type(8)))  float  v8f;
typedef __attribute__((ext_vector_type(4)))  unsigned int v4u;
typedef __attribute__((ext_vector_type(8)))  int v8i;
typedef __attribute__((ext_vector_type(4)))  int v4i;
typedef __attribute__((address_space(1))) v4i gv4i;   // global 16B packet
typedef __attribute__((address_space(3))) v4i lv4i;   // LDS 16B packet

#if defined(__has_builtin)
#if __has_builtin(__builtin_amdgcn_tensor_load_to_lds) && \
    __has_builtin(__builtin_amdgcn_s_wait_tensorcnt)
#define HAS_TDM 1
#endif
#if __has_builtin(__builtin_amdgcn_global_load_async_to_lds_b128) && \
    __has_builtin(__builtin_amdgcn_s_wait_asynccnt)
#define HAS_ASYNC 1
#endif
#endif
#ifndef HAS_TDM
#define HAS_TDM 0
#endif
#ifndef HAS_ASYNC
#define HAS_ASYNC 0
#endif

static constexpr int Bn = 2, Sn = 1024, Hn = 32, Dn = 128, EHn = 8;

static __device__ __forceinline__ v16bf cat16(v8bf lo, v8bf hi) {
  return __builtin_shufflevector(lo, hi, 0,1,2,3,4,5,6,7,8,9,10,11,12,13,14,15);
}
static __device__ __forceinline__ v8f wmma_bf16(v16bf a, v16bf b, v8f c) {
  // D = A(16x32) * B(32x16) + C(16x16 f32)
  return __builtin_amdgcn_wmma_f32_16x16x32_bf16(false, a, false, b, (short)0, c,
                                                 false, false);
}

// ---------------------------------------------------------------------------
// fp32 -> bf16 conversion
// ---------------------------------------------------------------------------
__global__ void f2bf_kernel(const float* __restrict__ in,
                            bf16_t* __restrict__ out, long long n) {
  long long i = ((long long)blockIdx.x * blockDim.x + threadIdx.x) * 8;
#pragma unroll
  for (int j = 0; j < 8; ++j)
    if (i + j < n) out[i + j] = (bf16_t)in[i + j];
}

// ---------------------------------------------------------------------------
// Generic bf16 WMMA GEMM: C[M,N] = A[M,K] * B[K,N]  (row-major)
// 256 threads = 8 waves; block tile 128x128; each wave 32x64 (2x4 WMMA tiles).
// ---------------------------------------------------------------------------
template <bool BIAS, bool GELU, bool OUTBF>
__global__ __launch_bounds__(256) void gemm_bf16_kernel(
    const bf16_t* __restrict__ A, const bf16_t* __restrict__ Bm,
    const float* __restrict__ bias, void* __restrict__ Cp,
    int M, int N, int K) {
  __shared__ bf16_t As[128][40];   // A tile 128x32, padded stride
  __shared__ bf16_t Bt[128][40];   // B tile transposed: [n][k]
  const int tid  = threadIdx.x;
  const int lane = tid & 31, wave = tid >> 5;
  const int l16  = lane & 15, lh = lane >> 4;
  const int wm   = wave & 3,  wn = wave >> 2;        // 4x2 wave grid
  const int m0   = blockIdx.x * 128, n0 = blockIdx.y * 128;

  const v8f vz = {0.f, 0.f, 0.f, 0.f, 0.f, 0.f, 0.f, 0.f};
  v8f acc[2][4];
#pragma unroll
  for (int i = 0; i < 2; ++i)
#pragma unroll
    for (int j = 0; j < 4; ++j) acc[i][j] = vz;

  for (int k0 = 0; k0 < K; k0 += 32) {
    __syncthreads();
#pragma unroll
    for (int i = 0; i < 2; ++i) {
      int e = tid * 8 + i * 2048;           // 4096 elems per tile
      int r = e >> 5, c = e & 31;           // A tile coords
#if HAS_ASYNC
      __builtin_amdgcn_global_load_async_to_lds_b128(
          (gv4i*)(A + (size_t)(m0 + r) * K + k0 + c), (lv4i*)&As[r][c], 0, 0);
#else
      *(v8bf*)&As[r][c] = *(const v8bf*)(A + (size_t)(m0 + r) * K + k0 + c);
#endif
      int kr = e >> 7, nc = e & 127;        // B tile coords
      v8bf bv = *(const v8bf*)(Bm + (size_t)(k0 + kr) * N + n0 + nc);
#pragma unroll
      for (int j = 0; j < 8; ++j) Bt[nc + j][kr] = bv[j];
    }
    if (k0 + 32 < K) {                      // prefetch next K-slab
      __builtin_prefetch(A + (size_t)(m0 + (tid & 127)) * K + (k0 + 32), 0, 1);
      __builtin_prefetch(Bm + (size_t)(k0 + 32 + (tid >> 3)) * N + n0 + (tid & 7) * 16,
                         0, 1);
    }
#if HAS_ASYNC
    __builtin_amdgcn_s_wait_asynccnt(0);
#endif
    __syncthreads();

    v16bf af[2], bfr[4];
#pragma unroll
    for (int i = 0; i < 2; ++i) {
      int row = wm * 32 + i * 16 + l16;     // A frag: lane = M row
      int cb  = lh * 8;                     // K chunks: cb..cb+7, cb+16..cb+23
      af[i] = cat16(*(const v8bf*)&As[row][cb], *(const v8bf*)&As[row][cb + 16]);
    }
#pragma unroll
    for (int j = 0; j < 4; ++j) {
      int col = wn * 64 + j * 16 + l16;     // B frag: lane = N col
      int kb  = lh * 8;                     // K chunks: kb..kb+7, kb+16..kb+23
      bfr[j] = cat16(*(const v8bf*)&Bt[col][kb], *(const v8bf*)&Bt[col][kb + 16]);
    }
#pragma unroll
    for (int i = 0; i < 2; ++i)
#pragma unroll
      for (int j = 0; j < 4; ++j) acc[i][j] = wmma_bf16(af[i], bfr[j], acc[i][j]);
  }

  // epilogue: C layout VGPR r -> M = r + 8*(lane/16), N = lane%16
#pragma unroll
  for (int i = 0; i < 2; ++i)
#pragma unroll
    for (int j = 0; j < 4; ++j)
#pragma unroll
      for (int r = 0; r < 8; ++r) {
        int row = m0 + wm * 32 + i * 16 + r + 8 * lh;
        int col = n0 + wn * 64 + j * 16 + l16;
        float v = acc[i][j][r];
        if (BIAS) v += bias[col];
        if (GELU) v = 0.5f * v * (1.f + erff(v * 0.70710678118654752f));
        if (OUTBF)
          ((bf16_t*)Cp)[(size_t)row * N + col] = (bf16_t)v;
        else
          ((float*)Cp)[(size_t)row * N + col] = v;
      }
}

// ---------------------------------------------------------------------------
// Token-type select + RoPE.  Reads per-type Q/K/V (bf16), writes [B,H,S,D] bf16.
// Head tiling: full head j uses source head j%8 for KV / entity tensors.
// ---------------------------------------------------------------------------
__global__ void rope_select_kernel(
    const bf16_t* __restrict__ qt,  const bf16_t* __restrict__ kt8,
    const bf16_t* __restrict__ vt8, const bf16_t* __restrict__ qe,
    const bf16_t* __restrict__ ke,  const bf16_t* __restrict__ ve,
    const int* __restrict__ ttm, const int* __restrict__ pos,
    bf16_t* __restrict__ Qo, bf16_t* __restrict__ Ko, bf16_t* __restrict__ Vo) {
  int idx = blockIdx.x * 256 + threadIdx.x;          // (b,s,h,d<64)
  if (idx >= Bn * Sn * Hn * (Dn / 2)) return;
  int d = idx & 63;
  int h = (idx >> 6) & 31;
  int s = (idx >> 11) & 1023;
  int b = idx >> 21;
  int bs = b * Sn + s;
  bool ise = (ttm[bs] == 1);
  float p = (float)pos[bs];
  // inv_freq = 10000^(-2d/128)
  float inv = __expf(-(float)(2 * d) * (9.210340371976184f / 128.f));
  float sn, cs;
  __sincosf(p * inv, &sn, &cs);

  size_t tbase = ((size_t)bs * Hn + h) * Dn;          // [B,S,32,128]
  size_t ebase = ((size_t)bs * EHn + (h & 7)) * Dn;   // [B,S,8,128]
  float q1, q2, k1, k2, w1, w2;
  if (ise) {
    q1 = (float)qe[ebase + d]; q2 = (float)qe[ebase + d + 64];
    k1 = (float)ke[ebase + d]; k2 = (float)ke[ebase + d + 64];
    w1 = (float)ve[ebase + d]; w2 = (float)ve[ebase + d + 64];
  } else {
    q1 = (float)qt[tbase + d]; q2 = (float)qt[tbase + d + 64];
    k1 = (float)kt8[ebase + d]; k2 = (float)kt8[ebase + d + 64];
    w1 = (float)vt8[ebase + d]; w2 = (float)vt8[ebase + d + 64];
  }
  size_t ob = (((size_t)b * Hn + h) * Sn + s) * Dn;   // [B,H,S,D]
  Qo[ob + d]      = (bf16_t)(q1 * cs - q2 * sn);
  Qo[ob + d + 64] = (bf16_t)(q2 * cs + q1 * sn);
  Ko[ob + d]      = (bf16_t)(k1 * cs - k2 * sn);
  Ko[ob + d + 64] = (bf16_t)(k2 * cs + k1 * sn);
  Vo[ob + d]      = (bf16_t)w1;
  Vo[ob + d + 64] = (bf16_t)w2;
}

// ---------------------------------------------------------------------------
// Flash attention: grid (S/128, H, B); 8 waves x 16 q-rows; causal mask -1e9.
// K tile staged by the Tensor Data Mover (padded LDS rows) when available.
// Online softmax in f32; S and P*V via WMMA; out bf16 [B,S,H,D].
// ---------------------------------------------------------------------------
__global__ __launch_bounds__(256) void flash_attn_kernel(
    const bf16_t* __restrict__ Qg, const bf16_t* __restrict__ Kg,
    const bf16_t* __restrict__ Vg, bf16_t* __restrict__ Og) {
  __shared__ bf16_t Ks[32][136];      // K tile [k][d]; 64 DW row + 4 DW pad
  __shared__ bf16_t Vt[128][40];      // V tile transposed [d][k]
  __shared__ bf16_t Ps[8][16][40];    // per-wave P scratch [qrow][k]
  const int tid  = threadIdx.x;
  const int lane = tid & 31, wave = tid >> 5;
  const int l16  = lane & 15, lh = lane >> 4;
  const int q0 = blockIdx.x * 128;
  const int h  = blockIdx.y, b = blockIdx.z;
  const size_t bh = ((size_t)b * Hn + h) * (size_t)Sn * Dn;
  const int qw = q0 + wave * 16;

#if HAS_TDM
  const unsigned ksbase =
      (unsigned)(unsigned long long)(__attribute__((address_space(3))) void*)&Ks[0][0];
  const unsigned long long kgbase = (unsigned long long)(const void*)(Kg + bh);
#endif

  // Q fragments: 4 x (16x32) over D=128, straight from global (layout matches)
  v16bf qf[4];
  {
    const bf16_t* qrow = Qg + bh + (size_t)(qw + l16) * Dn;
#pragma unroll
    for (int t = 0; t < 4; ++t) {
      int db = t * 32 + lh * 8;
      qf[t] = cat16(*(const v8bf*)(qrow + db), *(const v8bf*)(qrow + db + 16));
    }
  }
  const v8f vz = {0.f, 0.f, 0.f, 0.f, 0.f, 0.f, 0.f, 0.f};
  v8f o[8];
#pragma unroll
  for (int j = 0; j < 8; ++j) o[j] = vz;
  float mrow[8], lrow[8];
#pragma unroll
  for (int r = 0; r < 8; ++r) { mrow[r] = -1e30f; lrow[r] = 0.f; }

  const int ntiles = q0 / 32 + 4;     // causal window for this block
  for (int kt = 0; kt < ntiles; ++kt) {
    __syncthreads();
#if HAS_TDM
    if (tid < 32) {                   // wave 0 issues the TDM descriptor
      unsigned long long ga = kgbase + (unsigned long long)(kt * 32) * (Dn * 2);
      v4u g0 = {1u,                                   // count=1 (valid D#)
                ksbase,                               // lds_addr
                (unsigned)ga,                         // global_addr[31:0]
                (unsigned)((ga >> 32) & 0x1FFFFFFu) | (2u << 30)}; // addr[56:32]|type=2
      v8i g1 = {(int)((1u << 16) |                    // data_size = 2 bytes
                      (1u << 20) |                    // pad_enable
                      (5u << 22) |                    // pad_interval: 64 DWORDs
                      (3u << 25)),                    // pad_amount: 4 DWORDs
                (int)(128u << 16),                    // tensor_dim0 = 128 (lo16)
                (int)(32u << 16),                     // dim0 hi=0 | tensor_dim1 = 32
                (int)(128u << 16),                    // dim1 hi=0 | tile_dim0 = 128
                32,                                   // tile_dim1=32, tile_dim2=0
                128,                                  // tensor_dim0_stride = 128
                0, 0};                                // dim1_stride unused (2D)
      v4i gz = {0, 0, 0, 0};
#if __clang_major__ >= 23
      v8i gz8 = {0, 0, 0, 0, 0, 0, 0, 0};
      __builtin_amdgcn_tensor_load_to_lds(g0, g1, gz, gz, gz8, 0);
#else
      __builtin_amdgcn_tensor_load_to_lds(g0, g1, gz, gz, 0);
#endif
      __builtin_amdgcn_s_wait_tensorcnt(0);
    }
#endif
#pragma unroll
    for (int i = 0; i < 2; ++i) {     // stage V transposed (and K if no TDM)
      int e = tid * 8 + i * 2048;
      int kr = e >> 7, dc = e & 127;
      const size_t g = bh + (size_t)(kt * 32 + kr) * Dn + dc;
#if !HAS_TDM
      *(v8bf*)&Ks[kr][dc] = *(const v8bf*)(Kg + g);
#endif
      v8bf vv = *(const v8bf*)(Vg + g);
#pragma unroll
      for (int j = 0; j < 8; ++j) Vt[dc + j][kr] = vv[j];
    }
    if (kt + 1 < ntiles) {            // prefetch next K/V tile
      const size_t gp =
          bh + (size_t)((kt + 1) * 32 + (tid >> 3)) * Dn + (tid & 7) * 16;
      __builtin_prefetch(Kg + gp, 0, 1);
      __builtin_prefetch(Vg + gp, 0, 1);
    }
    __syncthreads();

    // scores S[16 x 32] = Q(16x128) . K^T, two 16x16 WMMA accumulators
    v8f s0 = vz, s1 = vz;
#pragma unroll
    for (int t = 0; t < 4; ++t) {
      int db = t * 32 + lh * 8;       // B frag: lane=k-col, d chunks db..db+7, db+16..db+23
      v16bf k0f = cat16(*(const v8bf*)&Ks[l16][db],
                        *(const v8bf*)&Ks[l16][db + 16]);
      v16bf k1f = cat16(*(const v8bf*)&Ks[l16 + 16][db],
                        *(const v8bf*)&Ks[l16 + 16][db + 16]);
      s0 = wmma_bf16(qf[t], k0f, s0);
      s1 = wmma_bf16(qf[t], k1f, s1);
    }

    const float scale = 0.08838834764831845f;   // 1/sqrt(128)
    float sc[8];
#pragma unroll
    for (int r = 0; r < 8; ++r) {
      int qi = qw + r + 8 * lh;
      int kj = kt * 32 + l16;
      float a = s0[r] * scale + ((kj > qi) ? -1e9f : 0.f);
      float c = s1[r] * scale + ((kj + 16 > qi) ? -1e9f : 0.f);
      float mx = fmaxf(a, c);
#pragma unroll
      for (int off = 1; off < 16; off <<= 1)
        mx = fmaxf(mx, __shfl_xor(mx, off, 32));
      float mo = mrow[r];
      float mn = fmaxf(mo, mx);
      float p0 = __expf(a - mn), p1 = __expf(c - mn);
      float ps = p0 + p1;
#pragma unroll
      for (int off = 1; off < 16; off <<= 1) ps += __shfl_xor(ps, off, 32);
      float esc = __expf(mo - mn);
      lrow[r] = lrow[r] * esc + ps;
      mrow[r] = mn;
      sc[r] = esc;
      Ps[wave][r + 8 * lh][l16]      = (bf16_t)p0;   // C-layout -> row-major
      Ps[wave][r + 8 * lh][l16 + 16] = (bf16_t)p1;
    }
#pragma unroll
    for (int j = 0; j < 8; ++j)
#pragma unroll
      for (int r = 0; r < 8; ++r) o[j][r] *= sc[r];

    // wave-private LDS round trip: ensure stores land before A-frag reads
    asm volatile("s_wait_dscnt 0" ::: "memory");
    v16bf pf = cat16(*(const v8bf*)&Ps[wave][l16][lh * 8],
                     *(const v8bf*)&Ps[wave][l16][lh * 8 + 16]);
#pragma unroll
    for (int j = 0; j < 8; ++j) {     // O += P(16x32) . V(32x16) over 8 d-tiles
      v16bf vf = cat16(*(const v8bf*)&Vt[j * 16 + l16][lh * 8],
                       *(const v8bf*)&Vt[j * 16 + l16][lh * 8 + 16]);
      o[j] = wmma_bf16(pf, vf, o[j]);
    }
  }

  // normalize and write [B,S,H,D] bf16
#pragma unroll
  for (int j = 0; j < 8; ++j)
#pragma unroll
    for (int r = 0; r < 8; ++r) {
      int qi = qw + r + 8 * lh;
      int dc = j * 16 + l16;
      float val = o[j][r] / lrow[r];
      Og[((size_t)(b * Sn + qi) * Hn + h) * Dn + dc] = (bf16_t)val;
    }
}

// ---------------------------------------------------------------------------
// oe[b,s,i,d] = mean over r of attn[b,s,r*8+i,d]   (r = 0..3)
// ---------------------------------------------------------------------------
__global__ void head_mean_kernel(const bf16_t* __restrict__ att,
                                 bf16_t* __restrict__ oe) {
  int idx = blockIdx.x * 256 + threadIdx.x;
  if (idx >= Bn * Sn * EHn * Dn) return;
  int d = idx & 127;
  int i = (idx >> 7) & 7;
  int bs = idx >> 10;
  float a = 0.f;
#pragma unroll
  for (int r = 0; r < 4; ++r)
    a += (float)att[((size_t)bs * Hn + r * 8 + i) * Dn + d];
  oe[idx] = (bf16_t)(a * 0.25f);
}

// ---------------------------------------------------------------------------
// In-place token-type masking of the f32 outputs
// ---------------------------------------------------------------------------
__global__ void apply_mask_kernel(float* __restrict__ p,
                                  const int* __restrict__ ttm,
                                  int hid, int keep_entity, long long n) {
  long long i = (long long)blockIdx.x * 256 + threadIdx.x;
  if (i >= n) return;
  int m = ttm[i / hid];
  float f = keep_entity ? (m == 1 ? 1.f : 0.f) : (m == 1 ? 0.f : 1.f);
  p[i] *= f;
}

// ---------------------------------------------------------------------------
// Host-side orchestration
// ---------------------------------------------------------------------------
extern "C" void kernel_launch(void* const* d_in, const int* in_sizes, int n_in,
                              void* d_out, int out_size, void* d_ws,
                              size_t ws_size, hipStream_t stream) {
  (void)in_sizes; (void)n_in; (void)out_size; (void)ws_size;
  const float* x_text   = (const float*)d_in[0];
  const float* x_entity = (const float*)d_in[1];
  const float* wq_t = (const float*)d_in[2];
  const float* wk_t = (const float*)d_in[3];
  const float* wv_t = (const float*)d_in[4];
  const float* wo_t = (const float*)d_in[5];
  const float* wq_e = (const float*)d_in[6];
  const float* wk_e = (const float*)d_in[7];
  const float* wv_e = (const float*)d_in[8];
  const float* wo_e = (const float*)d_in[9];
  const float* aq1_w = (const float*)d_in[10]; const float* aq1_b = (const float*)d_in[11];
  const float* aq2_w = (const float*)d_in[12]; const float* aq2_b = (const float*)d_in[13];
  const float* ak1_w = (const float*)d_in[14]; const float* ak1_b = (const float*)d_in[15];
  const float* ak2_w = (const float*)d_in[16]; const float* ak2_b = (const float*)d_in[17];
  const float* av1_w = (const float*)d_in[18]; const float* av1_b = (const float*)d_in[19];
  const float* av2_w = (const float*)d_in[20]; const float* av2_b = (const float*)d_in[21];
  const float* ao1_w = (const float*)d_in[22]; const float* ao1_b = (const float*)d_in[23];
  const float* ao2_w = (const float*)d_in[24]; const float* ao2_b = (const float*)d_in[25];
  // d_in[26] = attention_mask: pure causal -1e9, reproduced analytically
  const int* ttm = (const int*)d_in[27];
  const int* pos = (const int*)d_in[28];

  char* ws = (char*)d_ws;
  size_t off = 0;
  auto alloc = [&](size_t bytes) -> void* {
    void* p = ws + off;
    off += (bytes + 255) & ~(size_t)255;
    return p;
  };

  bf16_t* xt_bf  = (bf16_t*)alloc(2048ll * 4096 * 2);
  bf16_t* xe_bf  = (bf16_t*)alloc(2048ll * 1024 * 2);
  bf16_t* wqt_bf = (bf16_t*)alloc(4096ll * 4096 * 2);
  bf16_t* wkt_bf = (bf16_t*)alloc(4096ll * 1024 * 2);
  bf16_t* wvt_bf = (bf16_t*)alloc(4096ll * 1024 * 2);
  bf16_t* wot_bf = (bf16_t*)alloc(4096ll * 4096 * 2);
  bf16_t* wqe_bf = (bf16_t*)alloc(1024ll * 1024 * 2);
  bf16_t* wke_bf = (bf16_t*)alloc(1024ll * 1024 * 2);
  bf16_t* wve_bf = (bf16_t*)alloc(1024ll * 1024 * 2);
  bf16_t* woe_bf = (bf16_t*)alloc(1024ll * 1024 * 2);
  bf16_t* aq1 = (bf16_t*)alloc(128 * 256 * 2);
  bf16_t* aq2 = (bf16_t*)alloc(256 * 128 * 2);
  bf16_t* ak1 = (bf16_t*)alloc(128 * 256 * 2);
  bf16_t* ak2 = (bf16_t*)alloc(256 * 128 * 2);
  bf16_t* av1 = (bf16_t*)alloc(128 * 256 * 2);
  bf16_t* av2 = (bf16_t*)alloc(256 * 128 * 2);
  bf16_t* ao1 = (bf16_t*)alloc(128 * 256 * 2);
  bf16_t* ao2 = (bf16_t*)alloc(256 * 128 * 2);
  bf16_t* qt   = (bf16_t*)alloc(2048ll * 4096 * 2);
  bf16_t* kt8  = (bf16_t*)alloc(2048ll * 1024 * 2);
  bf16_t* vt8  = (bf16_t*)alloc(2048ll * 1024 * 2);
  bf16_t* qe0  = (bf16_t*)alloc(2048ll * 1024 * 2);   // later reused as qe
  bf16_t* ke0  = (bf16_t*)alloc(2048ll * 1024 * 2);   // later reused as ke
  bf16_t* ve0  = (bf16_t*)alloc(2048ll * 1024 * 2);   // later reused as ve
  bf16_t* hbuf = (bf16_t*)alloc(16384ll * 256 * 2);   // adapter hidden (serial reuse)
  bf16_t* Kr   = (bf16_t*)alloc((size_t)Bn * Hn * Sn * Dn * 2);
  bf16_t* Vr   = (bf16_t*)alloc((size_t)Bn * Hn * Sn * Dn * 2);
  bf16_t* oemn = (bf16_t*)alloc(16384ll * 128 * 2);   // mean, later reused as oe2
  bf16_t* Qr      = wqt_bf;   // dead after GEMM 1 (reconverted every call)
  bf16_t* attnout = xt_bf;    // dead after GEMMs 1-3

  auto conv = [&](const float* src, bf16_t* dst, long long n) {
    int blocks = (int)(((n + 7) / 8 + 255) / 256);
    f2bf_kernel<<<blocks, 256, 0, stream>>>(src, dst, n);
  };
  conv(x_text, xt_bf, 2048ll * 4096);
  conv(x_entity, xe_bf, 2048ll * 1024);
  conv(wq_t, wqt_bf, 4096ll * 4096);
  conv(wk_t, wkt_bf, 4096ll * 1024);
  conv(wv_t, wvt_bf, 4096ll * 1024);
  conv(wo_t, wot_bf, 4096ll * 4096);
  conv(wq_e, wqe_bf, 1024ll * 1024);
  conv(wk_e, wke_bf, 1024ll * 1024);
  conv(wv_e, wve_bf, 1024ll * 1024);
  conv(wo_e, woe_bf, 1024ll * 1024);
  conv(aq1_w, aq1, 128 * 256); conv(aq2_w, aq2, 256 * 128);
  conv(ak1_w, ak1, 128 * 256); conv(ak2_w, ak2, 256 * 128);
  conv(av1_w, av1, 128 * 256); conv(av2_w, av2, 256 * 128);
  conv(ao1_w, ao1, 128 * 256); conv(ao2_w, ao2, 256 * 128);

  const dim3 thr(256);
  // text projections
  gemm_bf16_kernel<false, false, true><<<dim3(16, 32), thr, 0, stream>>>(
      xt_bf, wqt_bf, nullptr, qt, 2048, 4096, 4096);
  gemm_bf16_kernel<false, false, true><<<dim3(16, 8), thr, 0, stream>>>(
      xt_bf, wkt_bf, nullptr, kt8, 2048, 1024, 4096);
  gemm_bf16_kernel<false, false, true><<<dim3(16, 8), thr, 0, stream>>>(
      xt_bf, wvt_bf, nullptr, vt8, 2048, 1024, 4096);
  // entity projections
  gemm_bf16_kernel<false, false, true><<<dim3(16, 8), thr, 0, stream>>>(
      xe_bf, wqe_bf, nullptr, qe0, 2048, 1024, 1024);
  gemm_bf16_kernel<false, false, true><<<dim3(16, 8), thr, 0, stream>>>(
      xe_bf, wke_bf, nullptr, ke0, 2048, 1024, 1024);
  gemm_bf16_kernel<false, false, true><<<dim3(16, 8), thr, 0, stream>>>(
      xe_bf, wve_bf, nullptr, ve0, 2048, 1024, 1024);
  // entity adapters: [16384,128] -> gelu([.,256]) -> [.,128]; out overwrites in
  gemm_bf16_kernel<true, true, true><<<dim3(128, 2), thr, 0, stream>>>(
      qe0, aq1, aq1_b, hbuf, 16384, 256, 128);
  gemm_bf16_kernel<true, false, true><<<dim3(128, 1), thr, 0, stream>>>(
      hbuf, aq2, aq2_b, qe0, 16384, 128, 256);
  gemm_bf16_kernel<true, true, true><<<dim3(128, 2), thr, 0, stream>>>(
      ke0, ak1, ak1_b, hbuf, 16384, 256, 128);
  gemm_bf16_kernel<true, false, true><<<dim3(128, 1), thr, 0, stream>>>(
      hbuf, ak2, ak2_b, ke0, 16384, 128, 256);
  gemm_bf16_kernel<true, true, true><<<dim3(128, 2), thr, 0, stream>>>(
      ve0, av1, av1_b, hbuf, 16384, 256, 128);
  gemm_bf16_kernel<true, false, true><<<dim3(128, 1), thr, 0, stream>>>(
      hbuf, av2, av2_b, ve0, 16384, 128, 256);
  // token-type select + RoPE -> [B,H,S,D]
  rope_select_kernel<<<16384, 256, 0, stream>>>(qt, kt8, vt8, qe0, ke0, ve0,
                                                ttm, pos, Qr, Kr, Vr);
  // flash attention -> [B,S,H,D] bf16
  flash_attn_kernel<<<dim3(Sn / 128, Hn, Bn), thr, 0, stream>>>(Qr, Kr, Vr,
                                                                attnout);
  // output projections
  float* out_text = (float*)d_out;
  float* out_ent  = (float*)d_out + 2048ll * 4096;
  gemm_bf16_kernel<false, false, false><<<dim3(16, 32), thr, 0, stream>>>(
      attnout, wot_bf, nullptr, out_text, 2048, 4096, 4096);
  head_mean_kernel<<<8192, 256, 0, stream>>>(attnout, oemn);
  gemm_bf16_kernel<true, true, true><<<dim3(128, 2), thr, 0, stream>>>(
      oemn, ao1, ao1_b, hbuf, 16384, 256, 128);
  gemm_bf16_kernel<true, false, true><<<dim3(128, 1), thr, 0, stream>>>(
      hbuf, ao2, ao2_b, oemn, 16384, 128, 256);
  gemm_bf16_kernel<false, false, false><<<dim3(16, 8), thr, 0, stream>>>(
      oemn, woe_bf, nullptr, out_ent, 2048, 1024, 1024);
  // token-type masking (in place on d_out)
  apply_mask_kernel<<<32768, 256, 0, stream>>>(out_text, ttm, 4096, 0,
                                               2048ll * 4096);
  apply_mask_kernel<<<8192, 256, 0, stream>>>(out_ent, ttm, 1024, 1,
                                              2048ll * 1024);
}
